// MultiQueryAttention_55868934586407
// MI455X (gfx1250) — matrix operand order repaired and checked by reading, not compiled, not run on hardware
//
#include <hip/hip_runtime.h>
#include <hip/hip_bf16.h>

// ---------------------------------------------------------------------------
// MQA attention for MI455X (gfx1250): bf16 WMMA everywhere, fp32 accumulate.
// Pipeline: [q@Wq *1/sqrt(hd)] -> Qbf ; [kv@Wkv] -> Kbf,Vbf ; flash-attn ->
// Obf ; [Obf@Wo + bo] -> out(fp32).
// K tiles are staged into LDS by the Tensor Data Mover (tensor_load_to_lds,
// with LDS padding matching the bank-conflict-free stride), synchronized via
// s_wait_tensorcnt.
// ---------------------------------------------------------------------------

typedef __attribute__((ext_vector_type(16))) __bf16 v16bf;
typedef __attribute__((ext_vector_type(8)))  __bf16 v8bf;
typedef __attribute__((ext_vector_type(8)))  float  v8f;
typedef __attribute__((ext_vector_type(4)))  float  v4f;
typedef __attribute__((ext_vector_type(4)))  unsigned u32x4;
typedef __attribute__((ext_vector_type(8)))  unsigned u32x8;

union V16 { v16bf v; v8bf h[2]; };

__device__ inline __bf16 f2bf(float f) {
  unsigned u = __builtin_bit_cast(unsigned, f);
  u += 0x7fffu + ((u >> 16) & 1u);           // round-to-nearest-even
  unsigned short s = (unsigned short)(u >> 16);
  return __builtin_bit_cast(__bf16, s);
}

// Cross-lane reductions within 16-lane halves via DS_SWIZZLE (group-of-32
// mode: offset[14:10]=xor, [4:0]=and). XOR masks < 16 stay inside a half.
template<int MASK>
__device__ inline float xmaxf(float x) {
  int i = __builtin_bit_cast(int, x);
  int j = __builtin_amdgcn_ds_swizzle(i, (MASK << 10) | 0x1f);
  return fmaxf(x, __builtin_bit_cast(float, j));
}
template<int MASK>
__device__ inline float xsumf(float x) {
  int i = __builtin_bit_cast(int, x);
  int j = __builtin_amdgcn_ds_swizzle(i, (MASK << 10) | 0x1f);
  return x + __builtin_bit_cast(float, j);
}

__device__ inline v8bf cvt8(const float* p) {
  v4f a = *(const v4f*)p;
  v4f b = *(const v4f*)(p + 4);
  v8bf r;
  r[0] = f2bf(a[0]); r[1] = f2bf(a[1]); r[2] = f2bf(a[2]); r[3] = f2bf(a[3]);
  r[4] = f2bf(b[0]); r[5] = f2bf(b[1]); r[6] = f2bf(b[2]); r[7] = f2bf(b[3]);
  return r;
}
__device__ inline v8bf cvt8(const __bf16* p) { return *(const v8bf*)p; }

__device__ inline void store_out(float*  p, float v) { *p = v; }
__device__ inline void store_out(__bf16* p, float v) { *p = f2bf(v); }

// ---------------------------------------------------------------------------
// Generic C = (A @ W + bias) * alpha   (A: MxK, W: KxN row-major fp32,
// out row-major).  128x128 tile / workgroup, 8 waves as 2(M) x 4(N),
// each wave 64x32 via 4x2 16x16 WMMA frags, K staged in 32-chunks.
// ---------------------------------------------------------------------------
template<typename AT, typename OT>
__global__ __launch_bounds__(256)
void gemm_bias_kernel(const AT* __restrict__ A, int lda,
                      const float* __restrict__ W, int ldw,
                      const float* __restrict__ bias,
                      OT* __restrict__ C, int ldc,
                      int K, float alpha) {
  __shared__ __bf16 Ash[128 * 40];   // 128 rows x 32 k (pad->40)
  __shared__ __bf16 Wt[128 * 40];    // W transposed: 128 n-rows x 32 k (pad->40)

  const int tid  = threadIdx.x;
  const int lane = tid & 31, wave = tid >> 5;
  const int hf   = lane >> 4, ln = lane & 15;
  const int wm   = wave >> 2, wn = wave & 3;
  const int m0   = blockIdx.y * 128, n0 = blockIdx.x * 128;

  v8f zero = {};
  v8f acc[4][2];
#pragma unroll
  for (int mf = 0; mf < 4; ++mf)
#pragma unroll
    for (int nf = 0; nf < 2; ++nf) acc[mf][nf] = zero;

  for (int k0 = 0; k0 < K; k0 += 32) {
    __syncthreads();
    // stage A tile (convert to bf16): 128x32, 8-elem groups
#pragma unroll
    for (int i = 0; i < 2; ++i) {
      int g = tid + i * 256;                  // 512 groups
      int row = g >> 2, col = (g & 3) * 8;
      *(v8bf*)&Ash[row * 40 + col] = cvt8(&A[(size_t)(m0 + row) * lda + k0 + col]);
    }
    // stage W transposed: read coalesced rows, write WtT[n][k]
#pragma unroll
    for (int i = 0; i < 4; ++i) {
      int g = tid + i * 256;                  // 1024 float4 groups
      int kk = g >> 5, nn = (g & 31) * 4;
      v4f w = *(const v4f*)&W[(size_t)(k0 + kk) * ldw + n0 + nn];
#pragma unroll
      for (int j = 0; j < 4; ++j) Wt[(nn + j) * 40 + kk] = f2bf(w[j]);
    }
    __syncthreads();

    V16 bfrag[2];
#pragma unroll
    for (int nf = 0; nf < 2; ++nf) {
      const __bf16* p = &Wt[(wn * 32 + nf * 16 + ln) * 40 + hf * 8];
      bfrag[nf].h[0] = *(const v8bf*)p;
      bfrag[nf].h[1] = *(const v8bf*)(p + 16);
    }
#pragma unroll
    for (int mf = 0; mf < 4; ++mf) {
      V16 af;
      const __bf16* p = &Ash[(wm * 64 + mf * 16 + ln) * 40 + hf * 8];
      af.h[0] = *(const v8bf*)p;          // K = 0..7  (+8*hf)
      af.h[1] = *(const v8bf*)(p + 16);   // K = 16..23 (+8*hf)
#pragma unroll
      for (int nf = 0; nf < 2; ++nf)
        acc[mf][nf] = __builtin_amdgcn_wmma_f32_16x16x32_bf16(
            false, af.v, false, bfrag[nf].v, (short)0, acc[mf][nf], false, false);
    }
  }

#pragma unroll
  for (int mf = 0; mf < 4; ++mf)
#pragma unroll
    for (int nf = 0; nf < 2; ++nf)
#pragma unroll
      for (int r = 0; r < 8; ++r) {
        int row = m0 + wm * 64 + mf * 16 + r + hf * 8;   // C layout: M = r + 8*half
        int col = n0 + wn * 32 + nf * 16 + ln;
        float v = (acc[mf][nf][r] + bias[col]) * alpha;
        store_out(&C[(size_t)row * ldc + col], v);
      }
}

// ---------------------------------------------------------------------------
// Flash attention.  Q: (2048,1024) bf16 pre-scaled by 1/sqrt(128).
// K,V: (2, 8192, 128) bf16 flat.  One WG per (b,h,128-row q-tile); 8 waves,
// each owns 16 q-rows.  KV chunk = 64 rows: 16 WMMA (QK^T) + 16 WMMA (PV).
// K chunk (contiguous 16KB) is DMA'd into LDS by the TDM with pad_enable
// inserting 16B after every 256B -> LDS row stride 136 elements.
// ---------------------------------------------------------------------------
__global__ __launch_bounds__(256)
void mqa_attn_kernel(const __bf16* __restrict__ Q,
                     const __bf16* __restrict__ Kb,
                     const __bf16* __restrict__ Vb,
                     __bf16* __restrict__ O) {
  __shared__ __bf16 Ksh[64 * 136];     // K rows (row-major, pad 128->136)
  __shared__ __bf16 Vt[128 * 72];      // V transposed: 128 d-rows x 64 kv (pad->72)
  __shared__ __bf16 Psh[8 * 16 * 72];  // per-wave P tiles, 16 x 64 (pad->72)

  const int tid  = threadIdx.x;
  const int lane = tid & 31, wave = tid >> 5;
  const int hf   = lane >> 4, ln = lane & 15;

  int idx = blockIdx.x;
  const int b  = idx >> 6; idx &= 63;
  const int h  = idx >> 3;
  const int qt = idx & 7;
  const int qrow_base = qt * 128 + wave * 16;        // q row within batch
  const size_t grow = (size_t)(b * 1024 + qrow_base + ln);

  // Load Q A-fragments once (d = 128 -> 4 frags of K=32)
  V16 aQ[4];
#pragma unroll
  for (int dc = 0; dc < 4; ++dc) {
    const __bf16* p = &Q[grow * 1024 + h * 128 + dc * 32 + hf * 8];
    aQ[dc].h[0] = *(const v8bf*)p;
    aQ[dc].h[1] = *(const v8bf*)(p + 16);
  }

  v8f zero = {};
  v8f acc[8];
#pragma unroll
  for (int of = 0; of < 8; ++of) acc[of] = zero;
  float m_r[8], l_r[8];
#pragma unroll
  for (int r = 0; r < 8; ++r) { m_r[r] = -3.0e38f; l_r[r] = 0.f; }

  const __bf16* Kbase = Kb + (size_t)b * (8192 * 128);
  const __bf16* Vbase = Vb + (size_t)b * (8192 * 128);
  __bf16* Pw = &Psh[wave * 16 * 72];

  // Loop-invariant part of the TDM descriptor (D# group 1):
  //   data_size=1 (2B), pad_enable, pad_interval=5 (64 DW = 256B = 128 elem),
  //   pad_amount=3 (4 DW = 16B = 8 elem), 1-D tile of 8192 elements.
  u32x8 g1;
  g1[0] = (1u << 16) | (1u << 20) | (5u << 22) | (3u << 25);
  g1[1] = 0u;                          // atomic_barrier_addr=0; tensor_dim0 lo16=0
  g1[2] = 0x10u | (1u << 16);          // tensor_dim0 = 0x100000; tensor_dim1 = 1
  g1[3] = 8192u << 16;                 // tile_dim0 = 8192 elements
  g1[4] = 0u;                          // tile_dim1 = tile_dim2 = 0 (unused)
  g1[5] = 0x100000u;                   // tensor_dim0_stride (unused for 1-D)
  g1[6] = 0u;
  g1[7] = 0u;
  const unsigned ksh_lds = (unsigned)(unsigned long long)(const void*)&Ksh[0];

  for (int kv0 = 0; kv0 < 8192; kv0 += 64) {
    __syncthreads();

    if (wave == 0) {
      // D# group 0: count=1, lds_addr, 57-bit global_addr, type=2 ("image")
      unsigned long long ga =
          (unsigned long long)(const void*)(Kbase + (size_t)kv0 * 128);
      u32x4 g0;
      g0[0] = 1u;
      g0[1] = ksh_lds;
      g0[2] = (unsigned)ga;
      g0[3] = ((unsigned)(ga >> 32) & 0x01FFFFFFu) | (2u << 30);
      asm volatile("tensor_load_to_lds %0, %1" :: "s"(g0), "s"(g1) : "memory");
    }

    // cooperative stage: V transposed (TDM cannot transpose)
#pragma unroll
    for (int i = 0; i < 4; ++i) {
      int g  = tid + i * 256;                 // 1024 groups of 8 bf16
      int kl = g >> 4, d8 = (g & 15) * 8;
      size_t goff = (size_t)(kv0 + kl) * 128 + d8;
      v8bf vvec = *(const v8bf*)&Vbase[goff];
#pragma unroll
      for (int j = 0; j < 8; ++j) Vt[(d8 + j) * 72 + kl] = vvec[j];
      if (kv0 + 64 < 8192)                    // prefetch next chunk
        __builtin_prefetch(&Vbase[goff + 64 * 128], 0, 1);
    }

    if (wave == 0) __builtin_amdgcn_s_wait_tensorcnt(0);
    __syncthreads();

    // S = Q @ K^T  (16 x 64)
    v8f s[4];
#pragma unroll
    for (int nf = 0; nf < 4; ++nf) {
      v8f c = zero;
#pragma unroll
      for (int dc = 0; dc < 4; ++dc) {
        V16 bK;
        const __bf16* p = &Ksh[(nf * 16 + ln) * 136 + dc * 32 + hf * 8];
        bK.h[0] = *(const v8bf*)p;
        bK.h[1] = *(const v8bf*)(p + 16);
        c = __builtin_amdgcn_wmma_f32_16x16x32_bf16(
            false, aQ[dc].v, false, bK.v, (short)0, c, false, false);
      }
      s[nf] = c;
    }

    // Online softmax (row = r + 8*hf, cols striped over 16 lanes of the half)
    float corr[8], lpart[8];
#pragma unroll
    for (int r = 0; r < 8; ++r) {
      float m = fmaxf(fmaxf(s[0][r], s[1][r]), fmaxf(s[2][r], s[3][r]));
      m = xmaxf<1>(m); m = xmaxf<2>(m); m = xmaxf<4>(m); m = xmaxf<8>(m);
      float mn = fmaxf(m_r[r], m);
      corr[r]  = __expf(m_r[r] - mn);
      m_r[r]   = mn;
      lpart[r] = 0.f;
    }
#pragma unroll
    for (int nf = 0; nf < 4; ++nf)
#pragma unroll
      for (int r = 0; r < 8; ++r) {
        float p = __expf(s[nf][r] - m_r[r]);
        s[nf][r] = p;
        lpart[r] += p;
      }
#pragma unroll
    for (int r = 0; r < 8; ++r) {
      float lp = lpart[r];
      lp = xsumf<1>(lp); lp = xsumf<2>(lp); lp = xsumf<4>(lp); lp = xsumf<8>(lp);
      l_r[r] = l_r[r] * corr[r] + lp;
    }
#pragma unroll
    for (int of = 0; of < 8; ++of)
#pragma unroll
      for (int r = 0; r < 8; ++r) acc[of][r] *= corr[r];

    // P (bf16) -> wave-private LDS, re-read in A-fragment layout
#pragma unroll
    for (int nf = 0; nf < 4; ++nf)
#pragma unroll
      for (int r = 0; r < 8; ++r)
        Pw[(r + hf * 8) * 72 + nf * 16 + ln] = f2bf(s[nf][r]);

    // O += P @ V
#pragma unroll
    for (int kc = 0; kc < 2; ++kc) {
      V16 aP;
      const __bf16* p = &Pw[ln * 72 + kc * 32 + hf * 8];
      aP.h[0] = *(const v8bf*)p;
      aP.h[1] = *(const v8bf*)(p + 16);
#pragma unroll
      for (int of = 0; of < 8; ++of) {
        V16 bV;
        const __bf16* pv = &Vt[(of * 16 + ln) * 72 + kc * 32 + hf * 8];
        bV.h[0] = *(const v8bf*)pv;
        bV.h[1] = *(const v8bf*)(pv + 16);
        acc[of] = __builtin_amdgcn_wmma_f32_16x16x32_bf16(
            false, aP.v, false, bV.v, (short)0, acc[of], false, false);
      }
    }
  }

  // normalize + store (B,S,1024) with heads back in columns h*128..
#pragma unroll
  for (int of = 0; of < 8; ++of)
#pragma unroll
    for (int r = 0; r < 8; ++r) {
      int row = b * 1024 + qrow_base + r + hf * 8;
      int col = h * 128 + of * 16 + ln;
      O[(size_t)row * 1024 + col] = f2bf(acc[of][r] / l_r[r]);
    }
}

// ---------------------------------------------------------------------------
extern "C" void kernel_launch(void* const* d_in, const int* in_sizes, int n_in,
                              void* d_out, int out_size, void* d_ws, size_t ws_size,
                              hipStream_t stream) {
  (void)in_sizes; (void)n_in; (void)out_size; (void)ws_size;
  const float* q   = (const float*)d_in[0];
  const float* kv  = (const float*)d_in[1];
  const float* Wq  = (const float*)d_in[2];
  const float* bq  = (const float*)d_in[3];
  const float* Wkv = (const float*)d_in[4];
  const float* bkv = (const float*)d_in[5];
  const float* Wo  = (const float*)d_in[6];
  const float* bo  = (const float*)d_in[7];
  float* out = (float*)d_out;

  __bf16* Qbf = (__bf16*)d_ws;                 // 2048x1024
  __bf16* Kbf = Qbf + (size_t)2048 * 1024;     // 2048x1024 == (2,8192,128)
  __bf16* Vbf = Kbf + (size_t)2048 * 1024;
  __bf16* Obf = Vbf + (size_t)2048 * 1024;     // total 16 MB workspace

  dim3 blk(256);
  dim3 gproj(8, 16);                           // (N/128, M/128) = (8,16)
  const float qscale = 0.08838834764831845f;   // 1/sqrt(128), folded into Q

  gemm_bias_kernel<float, __bf16><<<gproj, blk, 0, stream>>>(
      q, 1024, Wq, 1024, bq, Qbf, 1024, 1024, qscale);
  gemm_bias_kernel<float, __bf16><<<gproj, blk, 0, stream>>>(
      kv, 1024, Wkv, 2048, bkv, Kbf, 1024, 1024, 1.0f);
  gemm_bias_kernel<float, __bf16><<<gproj, blk, 0, stream>>>(
      kv, 1024, Wkv + 1024, 2048, bkv + 1024, Vbf, 1024, 1024, 1.0f);

  mqa_attn_kernel<<<dim3(128), blk, 0, stream>>>(Qbf, Kbf, Vbf, Obf);

  gemm_bias_kernel<__bf16, float><<<gproj, blk, 0, stream>>>(
      Obf, 1024, Wo, 1024, bo, out, 1024, 1024, 1.0f);
}
